// LM_35218731827864
// MI455X (gfx1250) — matrix operand order repaired and checked by reading, not compiled, not run on hardware
//
#include <hip/hip_runtime.h>

#define LSTM_D 1024
#define LSTM_H 1024
#define LSTM_B 64
#define LSTM_T 512
#define KSTEPS (LSTM_D / 32)   // 32 K-steps of 32
#define STAGES 4               // software-pipeline depth

typedef __attribute__((ext_vector_type(16))) __bf16 bf16x16;
typedef __attribute__((ext_vector_type(8)))  float  v8f;

union TileRegs {
  unsigned int u[8];
  uint4 q[2];
  bf16x16 v;
};

__device__ __forceinline__ unsigned short f2bf(float f) {
  unsigned int u = __builtin_bit_cast(unsigned int, f);
  u += 0x7FFFu + ((u >> 16) & 1u);   // round-to-nearest-even
  return (unsigned short)(u >> 16);
}

__device__ __forceinline__ unsigned int pk2(float a, float b) {
  return (unsigned int)f2bf(a) | ((unsigned int)f2bf(b) << 16);
}

__device__ __forceinline__ float sigm(float x) { return 1.0f / (1.0f + __expf(-x)); }

// ---------------- weight f32 -> bf16 conversion ----------------
__global__ void f32_to_bf16_kernel(const float* __restrict__ src,
                                   unsigned short* __restrict__ dst, int n) {
  for (int i = blockIdx.x * blockDim.x + threadIdx.x; i < n;
       i += gridDim.x * blockDim.x)
    dst[i] = f2bf(src[i]);
}

// ---------------- WMMA operand loaders (CDNA5 wave32 layouts) ----------------
// A 16x32 bf16: lanes 0-15 row=lane, K {0..7,16..23}; lanes 16-31 row=lane-16, K {8..15,24..31}
__device__ __forceinline__ void load_a_bf16(const unsigned short* __restrict__ base,
                                            int row, int kb, int lane, TileRegs& a) {
  const int klo = (lane & 16) ? 8 : 0;
  const unsigned short* p = base + (size_t)row * LSTM_D + kb + klo;
  a.q[0] = *(const uint4*)(p);        // K = kb+klo .. +7
  a.q[1] = *(const uint4*)(p + 16);   // K = kb+16+klo .. +7
}

// layer-1 A operand: embedding gather + f32->bf16 convert, emb row 0 forced to zero
__device__ __forceinline__ void load_a_emb(const float* __restrict__ emb,
                                           int token, int kb, int lane, TileRegs& a) {
  const float keep = (token == 0) ? 0.0f : 1.0f;
  const int klo = (lane & 16) ? 8 : 0;
  const float* p = emb + (size_t)token * LSTM_D + kb + klo;
  float4 f0 = *(const float4*)(p);
  float4 f1 = *(const float4*)(p + 4);
  float4 f2 = *(const float4*)(p + 16);
  float4 f3 = *(const float4*)(p + 20);
  a.u[0] = pk2(f0.x * keep, f0.y * keep);
  a.u[1] = pk2(f0.z * keep, f0.w * keep);
  a.u[2] = pk2(f1.x * keep, f1.y * keep);
  a.u[3] = pk2(f1.z * keep, f1.w * keep);
  a.u[4] = pk2(f2.x * keep, f2.y * keep);
  a.u[5] = pk2(f2.z * keep, f2.w * keep);
  a.u[6] = pk2(f3.x * keep, f3.y * keep);
  a.u[7] = pk2(f3.z * keep, f3.w * keep);
}

// B 32x16 bf16: lanes 0-15 K {0..7,16..23}, lanes 16-31 K {8..15,24..31}; col = lane&15
__device__ __forceinline__ void load_b(const unsigned short* __restrict__ w,
                                       int col, int kb, int lane, TileRegs& b) {
  const int koff = (lane & 16) ? 8 : 0;
  const unsigned short* p = w + (size_t)col * LSTM_D + kb + koff;
  b.q[0] = *(const uint4*)(p);
  b.q[1] = *(const uint4*)(p + 16);
}

// ---------------- one LSTM layer-step tile, split-K across the block's 8 waves ----
// Block: wave w -> gate = w&3, half = w>>2 (0: x@W_ih^T, 1: h@W_hh^T), K = 1024 each.
__device__ __forceinline__ void layer_body(
    const float* __restrict__ emb, const int* __restrict__ tok,
    const unsigned short* __restrict__ xin_bf16,
    const unsigned short* __restrict__ hprev_bf16,
    const unsigned short* __restrict__ Wih, const unsigned short* __restrict__ Whh,
    const float* __restrict__ bih, const float* __restrict__ bhh,
    float* __restrict__ c_state, unsigned short* __restrict__ hout_bf16,
    float* __restrict__ out_f32, int blk, float (*red)[8][32]) {
  const int tid = threadIdx.x;
  const int lane = tid & 31;
  const int wave = tid >> 5;           // 0..7
  const int gate = wave & 3;
  const int half = wave >> 2;          // 0 = input projection, 1 = recurrent projection
  const int rowTile = blk & 3;         // 4 x 16 batch rows
  const int unitTile = blk >> 2;       // 64 x 16 hidden units
  const int n = lane & 15;
  const int unit = unitTile * 16 + n;
  const int arow = rowTile * 16 + n;

  v8f acc = {};
  const unsigned short* W = half ? Whh : Wih;
  const int col = gate * LSTM_H + unit;

  TileRegs a[STAGES], b[STAGES];

  if (half == 0 && emb) {
    const int token = tok[arow];
#pragma unroll
    for (int s = 0; s < STAGES - 1; ++s) {
      load_a_emb(emb, token, s * 32, lane, a[s]);
      load_b(W, col, s * 32, lane, b[s]);
    }
#pragma unroll
    for (int i = 0; i < KSTEPS; ++i) {
      const int ld = i + STAGES - 1;
      if (ld < KSTEPS) {
        load_a_emb(emb, token, ld * 32, lane, a[ld & (STAGES - 1)]);
        load_b(W, col, ld * 32, lane, b[ld & (STAGES - 1)]);
      }
      acc = __builtin_amdgcn_wmma_f32_16x16x32_bf16(
          false, a[i & (STAGES - 1)].v, false, b[i & (STAGES - 1)].v,
          (short)0, acc, false, false);
    }
  } else {
    const unsigned short* X = half ? hprev_bf16 : xin_bf16;
#pragma unroll
    for (int s = 0; s < STAGES - 1; ++s) {
      load_a_bf16(X, arow, s * 32, lane, a[s]);
      load_b(W, col, s * 32, lane, b[s]);
    }
#pragma unroll
    for (int i = 0; i < KSTEPS; ++i) {
      const int ld = i + STAGES - 1;
      if (ld < KSTEPS) {
        load_a_bf16(X, arow, ld * 32, lane, a[ld & (STAGES - 1)]);
        load_b(W, col, ld * 32, lane, b[ld & (STAGES - 1)]);
      }
      acc = __builtin_amdgcn_wmma_f32_16x16x32_bf16(
          false, a[i & (STAGES - 1)].v, false, b[i & (STAGES - 1)].v,
          (short)0, acc, false, false);
    }
  }

  // combine split-K partials through LDS
#pragma unroll
  for (int r = 0; r < 8; ++r) red[wave][r][lane] = acc[r];
  __syncthreads();

  // fused cell update: thread tid owns element (m = tid>>4, n = tid&15)
  const int m = tid >> 4;
  const int nn = tid & 15;
  const int rl = m & 7;                    // C-tile VGPR index
  const int ll = nn + ((m >> 3) << 4);     // C-tile lane (lanes 16-31 hold M 8..15)
  const int unit2 = unitTile * 16 + nn;
  const int brow = rowTile * 16 + m;

  float gv[4];
#pragma unroll
  for (int g = 0; g < 4; ++g)
    gv[g] = red[g][rl][ll] + red[g + 4][rl][ll] +
            bih[g * LSTM_H + unit2] + bhh[g * LSTM_H + unit2];

  const float gi = sigm(gv[0]);
  const float gf = sigm(gv[1]);
  const float gg = tanhf(gv[2]);
  const float go = sigm(gv[3]);
  const int idx = brow * LSTM_H + unit2;
  const float cnew = gf * c_state[idx] + gi * gg;
  c_state[idx] = cnew;
  const float hn = go * tanhf(cnew);
  hout_bf16[idx] = f2bf(hn);
  if (out_f32) out_f32[idx] = hn;
}

// ---------------- fused cross-layer pipelined step ----------------
// Launch s: blocks 0..255 run layer1 at step s; blocks 256..511 run layer2 at step s-1.
// Both depend only on the previous launch -> serial chain is 513 launches, not 1024.
struct StepArgs {
  const float* emb;
  const int* tok;                    // null => L1 inactive this launch
  const unsigned short* h1prev;
  const unsigned short* Wih1;
  const unsigned short* Whh1;
  const float* bih1;
  const float* bhh1;
  float* c1;
  unsigned short* h1out;
  const unsigned short* xin2;        // h1(s-1); null => L2 inactive this launch
  const unsigned short* h2prev;
  const unsigned short* Wih2;
  const unsigned short* Whh2;
  const float* bih2;
  const float* bhh2;
  float* c2;
  unsigned short* h2out;
  float* out;                        // d_out + (s-1)*B*H
};

__global__ __launch_bounds__(256) void lstm_fused_step_kernel(StepArgs p) {
  __shared__ float red[8][8][32];
  if (blockIdx.x < 256) {
    if (!p.tok) return;
    layer_body(p.emb, p.tok, nullptr, p.h1prev, p.Wih1, p.Whh1, p.bih1, p.bhh1,
               p.c1, p.h1out, nullptr, blockIdx.x, red);
  } else {
    if (!p.xin2) return;
    layer_body(nullptr, nullptr, p.xin2, p.h2prev, p.Wih2, p.Whh2, p.bih2, p.bhh2,
               p.c2, p.h2out, p.out, blockIdx.x - 256, red);
  }
}

// ---------------- host driver ----------------
extern "C" void kernel_launch(void* const* d_in, const int* in_sizes, int n_in,
                              void* d_out, int out_size, void* d_ws, size_t ws_size,
                              hipStream_t stream) {
  const int*   tokens = (const int*)d_in[0];   // (513, 64)
  const float* emb    = (const float*)d_in[1]; // (32000, 1024)
  const float* W_ih   = (const float*)d_in[2]; // (2, 4096, 1024)
  const float* W_hh   = (const float*)d_in[3]; // (2, 4096, 1024)
  const float* b_ih   = (const float*)d_in[4]; // (2, 4096)
  const float* b_hh   = (const float*)d_in[5]; // (2, 4096)
  float* out = (float*)d_out;                  // (512, 64, 1024)

  const size_t wElems = (size_t)2 * 4096 * 1024;  // per weight tensor (both layers)
  const size_t hElems = (size_t)LSTM_B * LSTM_H;
  const size_t need = 2 * wElems * 2 /*bf16 weights*/ +
                      4 * hElems * 2 /*h ping-pong*/ + 2 * hElems * 4 /*c*/;
  if (ws_size < need) return;

  unsigned short* Wih_bf = (unsigned short*)d_ws;
  unsigned short* Whh_bf = Wih_bf + wElems;
  unsigned short* h1b[2] = {Whh_bf + wElems, Whh_bf + wElems + hElems};
  unsigned short* h2b[2] = {h1b[1] + hElems, h1b[1] + 2 * hElems};
  float* c1 = (float*)(h2b[1] + hElems);
  float* c2 = c1 + hElems;

  f32_to_bf16_kernel<<<1024, 256, 0, stream>>>(W_ih, Wih_bf, (int)wElems);
  f32_to_bf16_kernel<<<1024, 256, 0, stream>>>(W_hh, Whh_bf, (int)wElems);

  // zero h (both ping-pong buffers) and c state: contiguous 1 MB region
  hipMemsetAsync(h1b[0], 0, 4 * hElems * 2 + 2 * hElems * 4, stream);

  for (int s = 0; s <= LSTM_T; ++s) {
    const int p = s & 1;
    StepArgs args;
    args.emb = emb;
    // L1 at step s (active for s < 512): reads h1[p], writes h1[p^1]
    args.tok    = (s < LSTM_T) ? tokens + (size_t)s * LSTM_B : nullptr;
    args.h1prev = h1b[p];
    args.Wih1 = Wih_bf;
    args.Whh1 = Whh_bf;
    args.bih1 = b_ih;
    args.bhh1 = b_hh;
    args.c1 = c1;
    args.h1out = h1b[p ^ 1];
    // L2 at step s-1 (active for s >= 1): reads xin=h1[p] (written by L1(s-1)),
    // reads h2[p^1], writes h2[p]
    args.xin2   = (s >= 1) ? h1b[p] : nullptr;
    args.h2prev = h2b[p ^ 1];
    args.Wih2 = Wih_bf + (size_t)4096 * 1024;
    args.Whh2 = Whh_bf + (size_t)4096 * 1024;
    args.bih2 = b_ih + 4096;
    args.bhh2 = b_hh + 4096;
    args.c2 = c2;
    args.h2out = h2b[p];
    args.out = (s >= 1) ? out + (size_t)(s - 1) * LSTM_B * LSTM_H : nullptr;
    lstm_fused_step_kernel<<<512, 256, 0, stream>>>(args);
  }
}
